// GCNEncoder_84911503442106
// MI455X (gfx1250) — matrix operand order repaired and checked by reading, not compiled, not run on hardware
//
#include <hip/hip_runtime.h>

typedef __attribute__((ext_vector_type(16))) _Float16 v16h;
typedef __attribute__((ext_vector_type(8)))  float    v8f;

#define THREADS 256

// ---------------- utility kernels ----------------

__global__ void k_fill_u32(unsigned* p, unsigned v, int n) {
  int i = blockIdx.x * blockDim.x + threadIdx.x;
  if (i < n) p[i] = v;
}

__global__ void k_zero_f32v4(float4* p, int n4) {
  int i = blockIdx.x * blockDim.x + threadIdx.x;
  if (i < n4) p[i] = make_float4(0.f, 0.f, 0.f, 0.f);
}

__global__ void k_count_deg(const int* __restrict__ dst, unsigned* __restrict__ deg, int E) {
  int e = blockIdx.x * blockDim.x + threadIdx.x;
  if (e < E) atomicAdd(&deg[dst[e]], 1u);
}

__global__ void k_dis(const unsigned* __restrict__ deg, float* __restrict__ dis, int n) {
  int i = blockIdx.x * blockDim.x + threadIdx.x;
  if (i < n) dis[i] = rsqrtf((float)deg[i]);   // deg >= 1 (self-loop), no zero case
}

// Pack weight matrix W [Kreal x Nreal] (row-major, f32) into per-wave B fragments,
// in contiguous k order (k = 16h + e), matching the A loads in k_gemm_wmma:
//   lanes 0-15 : N = lane,      K = ks*32 + e        (e = 0..15)
//   lanes 16-31: N = lane - 16, K = ks*32 + 16 + e
// frag index = ((nt*nk + ks)*32 + lane)*16 + e
__global__ void k_wfrag(const float* __restrict__ W, int Kreal, int Nreal,
                        int nk, int ntiles, _Float16* __restrict__ frag) {
  int tid = blockIdx.x * blockDim.x + threadIdx.x;
  int total = ntiles * nk * 32 * 16;
  if (tid >= total) return;
  int e    = tid & 15;
  int lane = (tid >> 4) & 31;
  int ks   = (tid >> 9) % nk;
  int nt   = tid / (512 * nk);
  int n = nt * 16 + (lane & 15);
  int k = ks * 32 + ((lane >> 4) << 4) + e;
  float v = (k < Kreal && n < Nreal) ? W[(size_t)k * Nreal + n] : 0.0f;
  frag[tid] = (_Float16)v;
}

// ---------------- WMMA GEMM: C[M x Ncols] = A[M x K] * Wfrag ----------------
// One wave per 16x16 output tile. A is f32 (converted to f16 per lane following
// the same contiguous k order as k_wfrag (k = 16h + e)), B comes pre-packed from k_wfrag.
// mtiles * 16 == M exactly (100000 = 6250 * 16).
__global__ void k_gemm_wmma(const float* __restrict__ A, int lda,
                            const _Float16* __restrict__ frag,
                            float* __restrict__ C, int ldc,
                            int nk, int ntiles, int mtiles) {
  int wid  = (blockIdx.x * blockDim.x + threadIdx.x) >> 5;  // wave-uniform
  int lane = threadIdx.x & 31;
  if (wid >= mtiles * ntiles) return;                       // whole-wave exit, EXEC stays full
  int mt = wid / ntiles;
  int nt = wid % ntiles;

  // A layout: lanes 0-15 -> M = lane,     K = 0..15
  //           lanes16-31 -> M = lane-16,  K = 16..31
  const float* arow = A + (size_t)(mt * 16 + (lane & 15)) * lda + ((lane >> 4) << 4);
  const v16h*  bf   = (const v16h*)frag;

  v8f acc = {};
  for (int ks = 0; ks < nk; ++ks) {
    const float* ap = arow + ks * 32;
    v16h a;
#pragma unroll
    for (int i = 0; i < 8; ++i) a[i]     = (_Float16)ap[i];
#pragma unroll
    for (int i = 0; i < 8; ++i) a[8 + i] = (_Float16)ap[8 + i];
    v16h b = bf[(nt * nk + ks) * 32 + lane];
    acc = __builtin_amdgcn_wmma_f32_16x16x32_f16(
        /*neg_a=*/false, a, /*neg_b=*/false, b,
        /*c_mod=*/(short)0, acc, /*reuse_a=*/false, /*reuse_b=*/false);
  }

  // C/D layout: VGPR r -> M = r (lanes 0-15) / r+8 (lanes 16-31), N = lane&15
  int col   = nt * 16 + (lane & 15);
  int rbase = mt * 16 + ((lane >> 4) << 3);
#pragma unroll
  for (int r = 0; r < 8; ++r)
    C[(size_t)(rbase + r) * ldc + col] = acc[r];
}

// ---------------- edge scatter: AGG[dst] += H[src] * dis[src]*dis[dst] ----------------
// One wave per edge: the H[src] row read is a contiguous burst and the atomic
// adds from the wave cover a contiguous run of the destination row, so L2
// transactions coalesce (vs. 32 scattered words with thread-per-edge).
__global__ void k_scatter(const int* __restrict__ src, const int* __restrict__ dst,
                          const float* __restrict__ dis,
                          const float* __restrict__ H, int ldh,
                          float* __restrict__ AGG, int ldagg,
                          int ncols, int E) {
  int wid  = (blockIdx.x * blockDim.x + threadIdx.x) >> 5;  // edge id, wave-uniform
  int lane = threadIdx.x & 31;
  if (wid >= E) return;
  int s = src[wid], d = dst[wid];
  float w = dis[s] * dis[d];
  const float* h = H + (size_t)s * ldh;
  float*       a = AGG + (size_t)d * ldagg;
  for (int c = lane; c < ncols; c += 32)
    atomicAdd(&a[c], h[c] * w);
}

// self-loop term + bias (+ optional relu)
__global__ void k_finalize(const float* __restrict__ H, int ldh,
                           const float* __restrict__ dis, const float* __restrict__ bias,
                           float* __restrict__ AGG, int ldagg,
                           int ncols, int n, int do_relu) {
  int t = blockIdx.x * blockDim.x + threadIdx.x;
  if (t >= n * ncols) return;
  int i = t / ncols, c = t % ncols;
  float dii = dis[i];
  float v = AGG[(size_t)i * ldagg + c] + H[(size_t)i * ldh + c] * dii * dii + bias[c];
  AGG[(size_t)i * ldagg + c] = do_relu ? fmaxf(v, 0.0f) : v;
}

// ---------------- launch ----------------

extern "C" void kernel_launch(void* const* d_in, const int* in_sizes, int n_in,
                              void* d_out_v, int out_size, void* d_ws, size_t ws_size,
                              hipStream_t stream) {
  const float* x  = (const float*)d_in[0];
  const int*   ei = (const int*)  d_in[1];
  const float* W1 = (const float*)d_in[2];
  const float* b1 = (const float*)d_in[3];
  const float* W2 = (const float*)d_in[4];
  const float* b2 = (const float*)d_in[5];
  float* out = (float*)d_out_v;

  const int FIN = 256, HID = 40, FOUT = 20;
  const int N = in_sizes[0] / FIN;     // 100000
  const int E = in_sizes[1] / 2;       // 3200000
  const int* srcp = ei;
  const int* dstp = ei + E;

  const int HIDP  = 48;   // padded hidden (3 N-tiles of 16) — GEMM1 output stride
  const int HIDP2 = 64;   // padded K for layer 2 (2 K-steps of 32)
  const int OUTP  = 32;   // padded output (2 N-tiles)
  const int NK1 = FIN / 32;    // 8
  const int NT1 = HIDP / 16;   // 3
  const int NK2 = HIDP2 / 32;  // 2
  const int NT2 = OUTP / 16;   // 2
  const int MT  = N / 16;      // 6250 (exact)

  // workspace carve (aligned to 256B)
  char* ws = (char*)d_ws;
  size_t off = 0;
  auto walloc = [&](size_t bytes) -> void* {
    void* p = (void*)(ws + off);
    off += (bytes + 255) & ~(size_t)255;
    return p;
  };
  unsigned* deg = (unsigned*)walloc((size_t)N * 4);
  float*    dis = (float*)   walloc((size_t)N * 4);
  _Float16* w1f = (_Float16*)walloc((size_t)NT1 * NK1 * 32 * 16 * 2);
  _Float16* w2f = (_Float16*)walloc((size_t)NT2 * NK2 * 32 * 16 * 2);
  float*    h1  = (float*)   walloc((size_t)N * HIDP  * 4);
  float*    agg1= (float*)   walloc((size_t)N * HIDP2 * 4);
  float*    h2  = (float*)   walloc((size_t)N * OUTP  * 4);

  auto cdiv = [](int a, int b) { return (a + b - 1) / b; };

  // degrees + norm
  k_fill_u32 <<<cdiv(N, THREADS), THREADS, 0, stream>>>(deg, 1u, N);
  k_count_deg<<<cdiv(E, THREADS), THREADS, 0, stream>>>(dstp, deg, E);
  k_dis      <<<cdiv(N, THREADS), THREADS, 0, stream>>>(deg, dis, N);

  // pack weights into WMMA B fragments
  int nf1 = NT1 * NK1 * 32 * 16;
  int nf2 = NT2 * NK2 * 32 * 16;
  k_wfrag<<<cdiv(nf1, THREADS), THREADS, 0, stream>>>(W1, FIN, HID,  NK1, NT1, w1f);
  k_wfrag<<<cdiv(nf2, THREADS), THREADS, 0, stream>>>(W2, HID, FOUT, NK2, NT2, w2f);

  // layer 1: h1 = x @ W1 (WMMA), scatter, self-loop + bias + relu
  int tiles1 = MT * NT1;
  k_gemm_wmma<<<cdiv(tiles1 * 32, THREADS), THREADS, 0, stream>>>(
      x, FIN, w1f, h1, HIDP, NK1, NT1, MT);
  int na1_4 = N * HIDP2 / 4;
  k_zero_f32v4<<<cdiv(na1_4, THREADS), THREADS, 0, stream>>>((float4*)agg1, na1_4);
  k_scatter <<<cdiv(E * 32, THREADS), THREADS, 0, stream>>>(
      srcp, dstp, dis, h1, HIDP, agg1, HIDP2, HID, E);
  k_finalize<<<cdiv(N * HID, THREADS), THREADS, 0, stream>>>(
      h1, HIDP, dis, b1, agg1, HIDP2, HID, N, /*relu=*/1);

  // layer 2: h2 = relu_agg @ W2 (WMMA, K padded to 64 with zeros), scatter to out
  int tiles2 = MT * NT2;
  k_gemm_wmma<<<cdiv(tiles2 * 32, THREADS), THREADS, 0, stream>>>(
      agg1, HIDP2, w2f, h2, OUTP, NK2, NT2, MT);
  int no_4 = N * FOUT / 4;
  k_zero_f32v4<<<cdiv(no_4, THREADS), THREADS, 0, stream>>>((float4*)out, no_4);
  k_scatter <<<cdiv(E * 32, THREADS), THREADS, 0, stream>>>(
      srcp, dstp, dis, h2, OUTP, out, FOUT, FOUT, E);
  k_finalize<<<cdiv(N * FOUT, THREADS), THREADS, 0, stream>>>(
      h2, OUTP, dis, b2, out, FOUT, FOUT, N, /*relu=*/0);
}
